// CandidateVoteModule_V3_26688926777525
// MI455X (gfx1250) — matrix operand order repaired and checked by reading, not compiled, not run on hardware
//
#include <hip/hip_runtime.h>
#include <math.h>

typedef _Float16 h16;
typedef __attribute__((ext_vector_type(16))) _Float16 v16h;
typedef __attribute__((ext_vector_type(8)))  _Float16 v8h;
typedef __attribute__((ext_vector_type(8)))  float    v8f;

#define BB 4
#define NN 16384
#define SS 1024
#define NSAMP 32
#define CIN 32
#define CC 256
#define ODIM 128
#define K0P 64        // 3 + 32 padded up to multiple of 32
#define KFP 288       // 256 + 32
#define MSA (BB*SS*NSAMP)   // 131072
#define MFP (BB*NN)         // 65536
#define RADF 0.2f
#define EPSBN 1e-5f
#define TN 8          // n-tiles (16 cols each) per wave in GEMM

// ---------------- utility kernels ----------------

__global__ void k_zero(float* p, int n) {
  int i = blockIdx.x * blockDim.x + threadIdx.x;
  if (i < n) p[i] = 0.0f;
}

__global__ void k_cvtw(const float* __restrict__ W, h16* __restrict__ Wp,
                       int Cout, int Cin, int Kp) {
  int i = blockIdx.x * blockDim.x + threadIdx.x;
  if (i >= Cout * Kp) return;
  int o = i / Kp, k = i % Kp;
  Wp[i] = (k < Cin) ? (h16)W[o * Cin + k] : (h16)0.0f;
}

// ------- farthest point sampling: whole per-batch state in 320KB LDS -------
// xyz (192KB) + running min-dist (64KB) live in LDS for all 1024 rounds.

__global__ void k_fps(const float* __restrict__ xyz, int* __restrict__ inds) {
  const int b = blockIdx.x;
  const int t = threadIdx.x;
  const int T = blockDim.x;           // 512
  const int lane = t & 31, wid = t >> 5;
  extern __shared__ char smem[];
  float* sx = (float*)smem;           // 3*NN floats
  float* sd = sx + 3 * NN;            // NN floats
  __shared__ float rv[16];
  __shared__ int   ri[16];
  __shared__ int   s_last;
  const float* px = xyz + (size_t)b * NN * 3;
  for (int i = t; i < 3 * NN; i += T) sx[i] = px[i];
  for (int i = t; i < NN; i += T) sd[i] = 1e10f;
  if (t == 0) { s_last = 0; inds[b * SS] = 0; }
  __syncthreads();
  for (int it = 1; it < SS; ++it) {
    int last = s_last;
    float lx = sx[last * 3 + 0], ly = sx[last * 3 + 1], lz = sx[last * 3 + 2];
    float bv = -1.0f; int bi = 0;
    for (int i = t; i < NN; i += T) {
      float dx = sx[i * 3 + 0] - lx, dy = sx[i * 3 + 1] - ly, dz = sx[i * 3 + 2] - lz;
      float d2 = dx * dx + dy * dy + dz * dz;
      float od = sd[i];
      float nd = od < d2 ? od : d2;
      sd[i] = nd;
      if (nd > bv) { bv = nd; bi = i; }   // strict > keeps lowest index
    }
    // wave32 shuffle argmax, (max, min-index) combine is assoc+comm
#pragma unroll
    for (int o = 16; o > 0; o >>= 1) {
      float v2 = __shfl_xor(bv, o, 32);
      int   i2 = __shfl_xor(bi, o, 32);
      if (v2 > bv || (v2 == bv && i2 < bi)) { bv = v2; bi = i2; }
    }
    if (lane == 0) { rv[wid] = bv; ri[wid] = bi; }
    __syncthreads();
    if (wid == 0) {
      float v = (lane < (T >> 5)) ? rv[lane] : -1.0f;
      int   j = (lane < (T >> 5)) ? ri[lane] : 0x7fffffff;
#pragma unroll
      for (int o = 8; o > 0; o >>= 1) {
        float v2 = __shfl_xor(v, o, 32);
        int   j2 = __shfl_xor(j, o, 32);
        if (v2 > v || (v2 == v && j2 < j)) { v = v2; j = j2; }
      }
      if (lane == 0) { s_last = j; inds[b * SS + it] = j; }
    }
    __syncthreads();
  }
}

__global__ void k_newxyz(const float* __restrict__ xyz, const int* __restrict__ inds,
                         float* __restrict__ nxyz) {
  int i = blockIdx.x * blockDim.x + threadIdx.x;   // b*S+s
  if (i >= BB * SS) return;
  int b = i / SS;
  int p = inds[i];
  const float* src = xyz + ((size_t)b * NN + p) * 3;
  nxyz[i * 3 + 0] = src[0];
  nxyz[i * 3 + 1] = src[1];
  nxyz[i * 3 + 2] = src[2];
}

// ball query: one wave per center; ballot + prefix-popcount keeps exact
// "first NSAMP within radius, in index order" semantics, 32 pts/step.
__global__ void k_ball(const float* __restrict__ xyz, const float* __restrict__ nxyz,
                       int* __restrict__ idx) {
  int w = (int)((blockIdx.x * blockDim.x + threadIdx.x) >> 5);  // center id
  int lane = threadIdx.x & 31;
  if (w >= BB * SS) return;     // wave-uniform
  int b = w / SS;
  const float* px = xyz + (size_t)b * NN * 3;
  float cx = nxyz[w * 3 + 0], cy = nxyz[w * 3 + 1], cz = nxyz[w * 3 + 2];
  int* out = idx + (size_t)w * NSAMP;
  int cnt = 0, first = 0;
  for (int base = 0; base < NN && cnt < NSAMP; base += 32) {
    int n = base + lane;
    float dx = px[n * 3 + 0] - cx, dy = px[n * 3 + 1] - cy, dz = px[n * 3 + 2] - cz;
    bool pred = (dx * dx + dy * dy + dz * dz) < (RADF * RADF);
    unsigned mask = (unsigned)__ballot(pred);
    if (cnt == 0 && mask) first = base + __builtin_ctz(mask);
    int pos = cnt + __builtin_popcount(mask & ((1u << lane) - 1u));
    if (pred && pos < NSAMP) out[pos] = n;
    cnt += __builtin_popcount(mask);
  }
  if (cnt < NSAMP)
    for (int k = cnt + lane; k < NSAMP; k += 32) out[k] = first;
}

// build grouped SA input: rows (b,s,j) x cols [3 xyz | 32 feat | pad->64], f16
__global__ void k_group(const float* __restrict__ xyz, const float* __restrict__ feat,
                        const float* __restrict__ nxyz, const int* __restrict__ idx,
                        h16* __restrict__ X0) {
  int r = blockIdx.x * blockDim.x + threadIdx.x;
  if (r >= MSA) return;
  int bs = r / NSAMP;
  int b = bs / SS;
  int p = idx[r];
  const float* px = xyz + ((size_t)b * NN + p) * 3;
  const float inv = 1.0f / RADF;
  h16* row = X0 + (size_t)r * K0P;
  row[0] = (h16)((px[0] - nxyz[bs * 3 + 0]) * inv);
  row[1] = (h16)((px[1] - nxyz[bs * 3 + 1]) * inv);
  row[2] = (h16)((px[2] - nxyz[bs * 3 + 2]) * inv);
  const float* pf = feat + (size_t)b * CIN * NN + p;
  for (int c = 0; c < CIN; ++c) row[3 + c] = (h16)pf[(size_t)c * NN];
  for (int c = 3 + CIN; c < K0P; ++c) row[c] = (h16)0.0f;
}

// --------- WMMA GEMM: Y[M,Nc] = X[M,K] * W[Nc,K]^T (f16 in, f32 acc) --------
// 256-thread block = 8 waves sharing one 128-col n-group. The W strip
// (128 x K halfs, padded rows) is staged once into LDS; B fragments are then
// low-latency ds_load_b128 reads, A streams from global (the real bandwidth
// term). Each wave owns a 16x128 output strip (8 WMMA accumulators).
// Fused BN sum/sumsq atomics in the epilogue. Requires (M/16) % 8 == 0.

__global__ void k_gemm(const h16* __restrict__ X, const h16* __restrict__ W,
                       h16* __restrict__ Y, int M, int K, int Nc,
                       const float* __restrict__ bias, float* __restrict__ stat) {
  extern __shared__ h16 sw[];                     // 128 x (K+8) halfs
  const int lane = threadIdx.x & 31;
  const int wid  = threadIdx.x >> 5;              // 0..7
  const int ngrp = Nc >> 7;                       // groups of 8 n-tiles
  const int ng   = blockIdx.x % ngrp;
  const int mg   = blockIdx.x / ngrp;
  const int nbase = ng << 7;
  const int ldp  = K + 8;                         // pad 4 dwords: banks 4*m
  // cooperative stage of the W strip into LDS (b128 granularity)
  {
    const int kc8 = K >> 3;
    const int chunks = 128 * kc8;
    for (int c = threadIdx.x; c < chunks; c += blockDim.x) {
      int row = c / kc8, kc = (c % kc8) << 3;
      *(v8h*)(sw + row * ldp + kc) =
          *(const v8h*)(W + (size_t)(nbase + row) * K + kc);
    }
  }
  __syncthreads();
  const int mt = (mg << 3) + wid;
  const int m  = lane & 15;
  const int hs = lane >> 4;
  const h16* ap = X + (size_t)((mt << 4) + m) * K + hs * 8;
  const h16* bp = sw + m * ldp + hs * 8;
  v8f acc[TN] = {};
  for (int kb = 0; kb < K; kb += 32) {
    union { v16h v; v8h h[2]; } A;
    A.h[0] = *(const v8h*)(ap + kb);        // K = kb + 8*hs + [0..7]
    A.h[1] = *(const v8h*)(ap + kb + 16);   // K = kb + 16 + 8*hs + [0..7]
#pragma unroll
    for (int t = 0; t < TN; ++t) {
      union { v16h v; v8h h[2]; } Bf;
      const h16* bt = bp + t * 16 * ldp + kb;
      Bf.h[0] = *(const v8h*)(bt);          // K = kb + 8*hs + [0..7]
      Bf.h[1] = *(const v8h*)(bt + 16);     // K = kb + 16 + 8*hs + [0..7]
      acc[t] = __builtin_amdgcn_wmma_f32_16x16x32_f16(false, A.v, false, Bf.v,
                                                      (short)0, acc[t], false, false);
    }
  }
  const int row0 = (mt << 4) + hs * 8;
#pragma unroll
  for (int t = 0; t < TN; ++t) {
    const int col = nbase + (t << 4) + m;   // all 8 values of this lane/tile
    float bv = bias ? bias[col] : 0.0f;     // land in channel `col`
    float s = 0.0f, q = 0.0f;
#pragma unroll
    for (int v = 0; v < 8; ++v) {
      float x = acc[t][v] + bv;
      Y[(size_t)(row0 + v) * Nc + col] = (h16)x;
      s += x; q += x * x;
    }
    if (stat) {
      atomicAdd(stat + col, s);
      atomicAdd(stat + Nc + col, q);
    }
  }
}

// normalize + affine + relu, 8 halfs (b128) per thread; C % 8 == 0
__global__ void k_bnrelu(h16* __restrict__ Y, const float* __restrict__ stat,
                         const float* __restrict__ g, const float* __restrict__ be,
                         int M, int C) {
  size_t i = ((size_t)blockIdx.x * blockDim.x + threadIdx.x) * 8;
  if (i >= (size_t)M * C) return;
  int colb = (int)(i % C);                 // contiguous channel block of 8
  float invM = 1.0f / (float)M;
  v8h y = *(const v8h*)(Y + i);
  v8h o;
#pragma unroll
  for (int e = 0; e < 8; ++e) {
    int col = colb + e;
    float mean = stat[col] * invM;
    float var  = stat[C + col] * invM - mean * mean;
    float sc   = rsqrtf(var + EPSBN) * g[col];
    float v = ((float)y[e] - mean) * sc + be[col];
    o[e] = (h16)(v > 0.0f ? v : 0.0f);
  }
  *(v8h*)(Y + i) = o;
}

__global__ void k_maxpool(const h16* __restrict__ Xin, h16* __restrict__ saf) {
  int i = blockIdx.x * blockDim.x + threadIdx.x;    // (b*S+s)*C + c
  if (i >= BB * SS * CC) return;
  int bs = i / CC, c = i % CC;
  const h16* p = Xin + (size_t)bs * NSAMP * CC + c;
  float mv = -1e30f;
  for (int j = 0; j < NSAMP; ++j) { float v = (float)p[(size_t)j * CC]; if (v > mv) mv = v; }
  saf[i] = (h16)mv;
}

// 3-NN inverse-squared-distance interpolation + concat features -> X1[M,288]
__global__ void k_interp(const float* __restrict__ xyz, const float* __restrict__ nxyz,
                         const h16* __restrict__ saf, const float* __restrict__ feat,
                         h16* __restrict__ X1) {
  int i = blockIdx.x * blockDim.x + threadIdx.x;   // b*N+n
  if (i >= MFP) return;
  int b = i / NN, n = i % NN;
  float qx = xyz[(size_t)i * 3 + 0], qy = xyz[(size_t)i * 3 + 1], qz = xyz[(size_t)i * 3 + 2];
  const float* cs = nxyz + (size_t)b * SS * 3;
  float d0 = 1e30f, d1 = 1e30f, d2 = 1e30f;
  int i0 = 0, i1 = 0, i2 = 0;
  for (int s = 0; s < SS; ++s) {
    float dx = cs[s * 3 + 0] - qx, dy = cs[s * 3 + 1] - qy, dz = cs[s * 3 + 2] - qz;
    float d = dx * dx + dy * dy + dz * dz;
    if (d < d0)      { d2 = d1; i2 = i1; d1 = d0; i1 = i0; d0 = d; i0 = s; }
    else if (d < d1) { d2 = d1; i2 = i1; d1 = d; i1 = s; }
    else if (d < d2) { d2 = d; i2 = s; }
  }
  float r0 = 1.0f / (d0 + 1e-8f), r1 = 1.0f / (d1 + 1e-8f), r2 = 1.0f / (d2 + 1e-8f);
  float rs = 1.0f / (r0 + r1 + r2);
  float w0 = r0 * rs, w1 = r1 * rs, w2 = r2 * rs;
  const h16* f0 = saf + ((size_t)b * SS + i0) * CC;
  const h16* f1 = saf + ((size_t)b * SS + i1) * CC;
  const h16* f2 = saf + ((size_t)b * SS + i2) * CC;
  h16* row = X1 + (size_t)i * KFP;
  for (int c = 0; c < CC; ++c)
    row[c] = (h16)(w0 * (float)f0[c] + w1 * (float)f1[c] + w2 * (float)f2[c]);
  const float* pf = feat + (size_t)b * CIN * NN + n;
  for (int c = 0; c < CIN; ++c) row[CC + c] = (h16)pf[(size_t)c * NN];
}

// ---------------- head tails ----------------

__global__ void k_offset(const h16* __restrict__ Hb, const float* __restrict__ w3,
                         const float* __restrict__ b3, const float* __restrict__ xyz,
                         float* __restrict__ out_vote, float* __restrict__ out_off) {
  int i = blockIdx.x * blockDim.x + threadIdx.x;
  if (i >= MFP) return;
  const h16* h = Hb + (size_t)i * CC;
  for (int o = 0; o < 3; ++o) {
    const float* w = w3 + o * CC;
    float acc = b3[o];
    for (int c = 0; c < CC; ++c) acc += w[c] * (float)h[c];
    out_off[(size_t)i * 3 + o]  = acc;
    out_vote[(size_t)i * 3 + o] = xyz[(size_t)i * 3 + o] + acc;
  }
}

__global__ void k_score(const h16* __restrict__ Hb, const float* __restrict__ w2,
                        const float* __restrict__ b2, float* __restrict__ out_s) {
  int i = blockIdx.x * blockDim.x + threadIdx.x;
  if (i >= MFP) return;
  const h16* h = Hb + (size_t)i * CC;
  float acc = b2[0];
  for (int c = 0; c < CC; ++c) acc += w2[c] * (float)h[c];
  out_s[i] = 1.0f / (1.0f + expf(-acc));
}

__global__ void k_vnorm(const h16* __restrict__ VF, float* __restrict__ out) {
  int i = blockIdx.x * blockDim.x + threadIdx.x;
  if (i >= MFP) return;
  const h16* v = VF + (size_t)i * ODIM;
  float s = 0.0f;
  for (int c = 0; c < ODIM; ++c) { float x = (float)v[c]; s += x * x; }
  float nrm = sqrtf(s);
  if (nrm < 1e-12f) nrm = 1e-12f;
  float inv = 1.0f / nrm;
  for (int c = 0; c < ODIM; ++c) out[(size_t)i * ODIM + c] = (float)v[c] * inv;
}

// ---------------- host ----------------

extern "C" void kernel_launch(void* const* d_in, const int* in_sizes, int n_in,
                              void* d_out, int out_size, void* d_ws, size_t ws_size,
                              hipStream_t stream) {
  (void)in_sizes; (void)n_in; (void)out_size; (void)ws_size;
  const float* xyz   = (const float*)d_in[0];
  const float* feat  = (const float*)d_in[1];
  const float* sa_w0 = (const float*)d_in[2];
  const float* sa_w1 = (const float*)d_in[3];
  const float* sa_w2 = (const float*)d_in[4];
  const float* sa_g0 = (const float*)d_in[5];
  const float* sa_b0 = (const float*)d_in[6];
  const float* sa_g1 = (const float*)d_in[7];
  const float* sa_b1 = (const float*)d_in[8];
  const float* sa_g2 = (const float*)d_in[9];
  const float* sa_b2 = (const float*)d_in[10];
  const float* fp_w0 = (const float*)d_in[11];
  const float* fp_w1 = (const float*)d_in[12];
  const float* fp_g0 = (const float*)d_in[13];
  const float* fp_b0 = (const float*)d_in[14];
  const float* fp_g1 = (const float*)d_in[15];
  const float* fp_b1 = (const float*)d_in[16];
  const float* off_w1  = (const float*)d_in[17];
  const float* off_g1  = (const float*)d_in[19];
  const float* off_be1 = (const float*)d_in[20];
  const float* off_w2  = (const float*)d_in[21];
  const float* off_g2  = (const float*)d_in[23];
  const float* off_be2 = (const float*)d_in[24];
  const float* off_w3  = (const float*)d_in[25];
  const float* off_b3  = (const float*)d_in[26];
  const float* sco_w1  = (const float*)d_in[27];
  const float* sco_g1  = (const float*)d_in[29];
  const float* sco_be1 = (const float*)d_in[30];
  const float* sco_w2  = (const float*)d_in[31];
  const float* sco_b2  = (const float*)d_in[32];
  const float* fea_w1  = (const float*)d_in[33];
  const float* fea_g1  = (const float*)d_in[35];
  const float* fea_be1 = (const float*)d_in[36];
  const float* fea_w2  = (const float*)d_in[37];
  const float* fea_g2  = (const float*)d_in[39];
  const float* fea_be2 = (const float*)d_in[40];
  const float* fea_w3  = (const float*)d_in[41];
  const float* fea_b3  = (const float*)d_in[42];
  // (conv biases preceding a training-mode BN cancel exactly; not used)

  char* base = (char*)d_ws;
  size_t off = 0;
  auto alloc = [&](size_t bytes) -> void* {
    void* r = base + off;
    off = (off + bytes + 255) & ~(size_t)255;
    return r;
  };
  int*   inds  = (int*)  alloc((size_t)BB * SS * 4);
  float* nxyz  = (float*)alloc((size_t)BB * SS * 3 * 4);
  int*   gidx  = (int*)  alloc((size_t)BB * SS * NSAMP * 4);
  float* stat  = (float*)alloc((size_t)10 * 2 * CC * 4);
  h16* w_sa0 = (h16*)alloc((size_t)CC * K0P * 2);
  h16* w_sa1 = (h16*)alloc((size_t)CC * CC * 2);
  h16* w_sa2 = (h16*)alloc((size_t)CC * CC * 2);
  h16* w_fp0 = (h16*)alloc((size_t)CC * KFP * 2);
  h16* w_fp1 = (h16*)alloc((size_t)CC * CC * 2);
  h16* w_of1 = (h16*)alloc((size_t)CC * CC * 2);
  h16* w_of2 = (h16*)alloc((size_t)CC * CC * 2);
  h16* w_sc1 = (h16*)alloc((size_t)CC * CC * 2);
  h16* w_fe1 = (h16*)alloc((size_t)CC * CC * 2);
  h16* w_fe2 = (h16*)alloc((size_t)CC * CC * 2);
  h16* w_fe3 = (h16*)alloc((size_t)ODIM * CC * 2);
  h16* saf = (h16*)alloc((size_t)BB * SS * CC * 2);
  h16* X0  = (h16*)alloc((size_t)MSA * K0P * 2);   // 16 MB, reused as VF later
  h16* A1  = (h16*)alloc((size_t)MSA * CC * 2);    // 64 MB
  h16* A2  = (h16*)alloc((size_t)MSA * CC * 2);    // 64 MB
  h16* X1  = (h16*)alloc((size_t)MFP * KFP * 2);   // 36 MB
  h16* B1  = A1;                                   // 32 MB region
  h16* B2  = A1 + (size_t)MFP * CC;                // 32 MB region
  h16* FPF = A2;                                   // persistent fp_feat
  h16* VF  = X0;                                   // 65536*128*2 == MSA*K0P*2

  auto cvt = [&](const float* W, h16* Wp, int Cout, int Cin, int Kp) {
    int n = Cout * Kp;
    k_cvtw<<<(n + 255) / 256, 256, 0, stream>>>(W, Wp, Cout, Cin, Kp);
  };
  auto gemm = [&](const h16* X, const h16* Wp, h16* Y, int M, int K, int Nc,
                  const float* bias, float* st) {
    int blocks = (M / 16 / 8) * (Nc / 128);         // 8 waves x 16x128 strips
    size_t smem = (size_t)128 * (K + 8) * sizeof(h16);
    k_gemm<<<blocks, 256, smem, stream>>>(X, Wp, Y, M, K, Nc, bias, st);
  };
  auto bnrelu = [&](h16* Y, float* st, const float* g, const float* be, int M) {
    size_t tot = (size_t)M * CC / 8;
    k_bnrelu<<<(unsigned)((tot + 255) / 256), 256, 0, stream>>>(Y, st, g, be, M, CC);
  };
  float* st = stat;  // 10 slots of 2*CC floats

  k_zero<<<(10 * 2 * CC + 255) / 256, 256, 0, stream>>>(stat, 10 * 2 * CC);
  cvt(sa_w0, w_sa0, CC, CIN + 3, K0P);
  cvt(sa_w1, w_sa1, CC, CC, CC);
  cvt(sa_w2, w_sa2, CC, CC, CC);
  cvt(fp_w0, w_fp0, CC, KFP, KFP);
  cvt(fp_w1, w_fp1, CC, CC, CC);
  cvt(off_w1, w_of1, CC, CC, CC);
  cvt(off_w2, w_of2, CC, CC, CC);
  cvt(sco_w1, w_sc1, CC, CC, CC);
  cvt(fea_w1, w_fe1, CC, CC, CC);
  cvt(fea_w2, w_fe2, CC, CC, CC);
  cvt(fea_w3, w_fe3, ODIM, CC, CC);

  // --- sampling & grouping ---
  k_fps<<<BB, 512, (3 * NN + NN) * sizeof(float), stream>>>(xyz, inds);
  k_newxyz<<<(BB * SS + 255) / 256, 256, 0, stream>>>(xyz, inds, nxyz);
  k_ball<<<(BB * SS * 32 + 255) / 256, 256, 0, stream>>>(xyz, nxyz, gidx);
  k_group<<<(MSA + 255) / 256, 256, 0, stream>>>(xyz, feat, nxyz, gidx, X0);

  // --- SA MLP (3 layers) + maxpool ---
  gemm(X0, w_sa0, A1, MSA, K0P, CC, nullptr, st + 0 * 2 * CC);
  bnrelu(A1, st + 0 * 2 * CC, sa_g0, sa_b0, MSA);
  gemm(A1, w_sa1, A2, MSA, CC, CC, nullptr, st + 1 * 2 * CC);
  bnrelu(A2, st + 1 * 2 * CC, sa_g1, sa_b1, MSA);
  gemm(A2, w_sa2, A1, MSA, CC, CC, nullptr, st + 2 * 2 * CC);
  bnrelu(A1, st + 2 * 2 * CC, sa_g2, sa_b2, MSA);
  k_maxpool<<<(BB * SS * CC + 255) / 256, 256, 0, stream>>>(A1, saf);

  // --- FP interpolation + MLP (2 layers) ---
  k_interp<<<(MFP + 127) / 128, 128, 0, stream>>>(xyz, nxyz, saf, feat, X1);
  gemm(X1, w_fp0, B1, MFP, KFP, CC, nullptr, st + 3 * 2 * CC);
  bnrelu(B1, st + 3 * 2 * CC, fp_g0, fp_b0, MFP);
  gemm(B1, w_fp1, FPF, MFP, CC, CC, nullptr, st + 4 * 2 * CC);
  bnrelu(FPF, st + 4 * 2 * CC, fp_g1, fp_b1, MFP);

  float* out       = (float*)d_out;
  float* out_vote  = out;
  float* out_off   = out + (size_t)BB * NN * 3;
  float* out_sco   = out + (size_t)2 * BB * NN * 3;
  float* out_vfeat = out + (size_t)2 * BB * NN * 3 + (size_t)BB * NN;

  // --- offset head ---
  gemm(FPF, w_of1, B1, MFP, CC, CC, nullptr, st + 5 * 2 * CC);
  bnrelu(B1, st + 5 * 2 * CC, off_g1, off_be1, MFP);
  gemm(B1, w_of2, B2, MFP, CC, CC, nullptr, st + 6 * 2 * CC);
  bnrelu(B2, st + 6 * 2 * CC, off_g2, off_be2, MFP);
  k_offset<<<(MFP + 127) / 128, 128, 0, stream>>>(B2, off_w3, off_b3, xyz, out_vote, out_off);

  // --- score head ---
  gemm(FPF, w_sc1, B1, MFP, CC, CC, nullptr, st + 7 * 2 * CC);
  bnrelu(B1, st + 7 * 2 * CC, sco_g1, sco_be1, MFP);
  k_score<<<(MFP + 127) / 128, 128, 0, stream>>>(B1, sco_w2, sco_b2, out_sco);

  // --- feature head ---
  gemm(FPF, w_fe1, B1, MFP, CC, CC, nullptr, st + 8 * 2 * CC);
  bnrelu(B1, st + 8 * 2 * CC, fea_g1, fea_be1, MFP);
  gemm(B1, w_fe2, B2, MFP, CC, CC, nullptr, st + 9 * 2 * CC);
  bnrelu(B2, st + 9 * 2 * CC, fea_g2, fea_be2, MFP);
  gemm(B2, w_fe3, VF, MFP, CC, ODIM, fea_b3, nullptr);
  k_vnorm<<<(MFP + 127) / 128, 128, 0, stream>>>(VF, out_vfeat);
}
